// DiffHead_38757784879894
// MI455X (gfx1250) — matrix operand order repaired and checked by reading, not compiled, not run on hardware
//
#include <hip/hip_runtime.h>
#include <math.h>

typedef __attribute__((ext_vector_type(16))) __bf16 v16bf;
typedef __attribute__((ext_vector_type(8)))  __bf16 v8bf;
typedef __attribute__((ext_vector_type(8)))  float  v8f;

#define DIM 256
#define NF  100
#define NROW 768           // B*NL = 256*3
#define RC  (DIM*NF)       // 25600 floats per (row)
#define BN_INV 0.99999500f // 1/sqrt(1+1e-5)

__device__ __forceinline__ float gelu_exact(float v) {
    return 0.5f * v * (1.0f + erff(v * 0.70710678118654752f));
}

// A-fragment K permutation for 16-bit WMMA (16x32 A): lanes 0-15 hold K {0..15},
// lanes 16-31 hold K {16..31}; element j maps as below (one contiguous 16-run).
__device__ __forceinline__ int a_perm(int j, int half) {
    return j + 16 * half;
}

// two aligned 16B LDS loads -> one 16-element bf16 fragment
__device__ __forceinline__ v16bf load_bfrag(const __bf16* p) {
    union { v16bf v; v8bf h[2]; } u;
    u.h[0] = *(const v8bf*)p;
    u.h[1] = *(const v8bf*)(p + 8);
    return u.v;
}

// ---------------------------------------------------------------------------
// weight pre-swizzle: GEMM weights (Cout x K) -> bf16 WMMA A-fragment order
// layout: frag[((mt*ksteps + ks)*32 + lane)*16 + j]
// ---------------------------------------------------------------------------
__global__ void wfrag_gemm_kernel(const float* __restrict__ w, __bf16* __restrict__ f,
                                  int Cout, int K)
{
    int ksteps = K / 32;
    int total = (Cout / 16) * ksteps * 32 * 16;
    for (int i = blockIdx.x * blockDim.x + threadIdx.x; i < total;
         i += gridDim.x * blockDim.x) {
        int j = i & 15;
        int lane = (i >> 4) & 31;
        int rest = i >> 9;
        int ks = rest % ksteps;
        int mt = rest / ksteps;
        int half = lane >> 4, lm = lane & 15;
        int o = mt * 16 + lm;
        int k = ks * 32 + a_perm(j, half);
        f[i] = (__bf16)w[(size_t)o * K + k];
    }
}

// conv weights (Cout,Cin,3,3), tap-major k = t*Cin + c, K zero-padded to 32
__global__ void wfrag_conv_kernel(const float* __restrict__ w, __bf16* __restrict__ f,
                                  int Cout, int Cin)
{
    int Ktot = Cin * 9;
    int ksteps = (Ktot + 31) / 32;
    int total = (Cout / 16) * ksteps * 32 * 16;
    for (int i = blockIdx.x * blockDim.x + threadIdx.x; i < total;
         i += gridDim.x * blockDim.x) {
        int j = i & 15;
        int lane = (i >> 4) & 31;
        int rest = i >> 9;
        int ks = rest % ksteps;
        int mt = rest / ksteps;
        int half = lane >> 4, lm = lane & 15;
        int o = mt * 16 + lm;
        int k = ks * 32 + a_perm(j, half);
        float v = 0.f;
        if (k < Ktot) {
            int t = k / Cin;
            int c = k - t * Cin;
            v = w[(size_t)(o * Cin + c) * 9 + t];
        }
        f[i] = (__bf16)v;
    }
}

// ---------------------------------------------------------------------------
// K1: depthwise tconv + 0.5*(relu(t)+x)  ->  xcat (= out_list) (768,256,100)
// ---------------------------------------------------------------------------
__global__ __launch_bounds__(256)
void tconv_kernel(const float* __restrict__ x0, const float* __restrict__ x1,
                  const float* __restrict__ x2, const float* __restrict__ tw,
                  const float* __restrict__ tb, float* __restrict__ xcat)
{
    int r = blockIdx.x;            // 0..767
    int layer = r % 3, b = r / 3;
    const float* src = (layer == 0) ? x0 : (layer == 1) ? x1 : x2;
    src += (size_t)b * RC;
    float* dst = xcat + (size_t)r * RC;
    for (int idx = threadIdx.x; idx < RC; idx += blockDim.x) {
        int c = idx / NF, l = idx % NF;
        const float* w = tw + ((size_t)layer * DIM + c) * 3;
        float xm = (l > 0)      ? src[c * NF + l - 1] : 0.f;
        float xc = src[c * NF + l];
        float xp = (l < NF - 1) ? src[c * NF + l + 1] : 0.f;
        float y = w[0] * xm + w[1] * xc + w[2] * xp + tb[layer * DIM + c];
        y = fmaxf(y, 0.f);
        dst[idx] = 0.5f * (y + xc);
    }
}

// ---------------------------------------------------------------------------
// K2: DiffMixer — x = xcat + gelu(bn(W0@xn+b0)) + gelu(bn(W1@dx+b1))
// grid (4, 768) block 256; N=32 per WG; bf16 n-major LDS tiles for xn and dx;
// pre-swizzled bf16 A frags; 8 WMMA per K-step per wave.
// ---------------------------------------------------------------------------
__global__ __launch_bounds__(256)
void diffmixer_kernel(const float* __restrict__ xcat,
                      const __bf16* __restrict__ a0f, const float* __restrict__ b0,
                      const float* __restrict__ g0, const float* __restrict__ be0,
                      const __bf16* __restrict__ a1f, const float* __restrict__ b1,
                      const float* __restrict__ g1, const float* __restrict__ be1,
                      const float* __restrict__ preg, const float* __restrict__ preb,
                      float* __restrict__ xout)
{
    __shared__ __attribute__((aligned(32))) __bf16 B0t[32][40];  // [col][k]
    __shared__ __attribute__((aligned(32))) __bf16 B1t[32][40];
    int r = blockIdx.y;
    int L0 = blockIdx.x * 32;
    int tid = threadIdx.x, lane = tid & 31, wave = tid >> 5;
    int lm = lane & 15, half = lane >> 4;
    const float* xr = xcat + (size_t)r * RC;

    v8f acc[2][2][2] = {};                    // [mi][mat][ni]
    for (int ks = 0; ks < 8; ++ks) {
        int k0 = ks * 32;
#pragma unroll
        for (int u = 0; u < 4; ++u) {
            int idx = tid + u * 256;          // 32 cols x 32 k
            int jcol = idx & 31, kk = idx >> 5;
            int c = k0 + kk;
            int l = L0 + jcol;
            int lcur = l > NF - 1 ? NF - 1 : l;
            int lprv = (l == 0) ? 1 : (l - 1 > NF - 1 ? NF - 1 : l - 1);
            float sc = preg[c] * BN_INV, sb = preb[c];
            float cur = xr[c * NF + lcur] * sc + sb;
            float prv = xr[c * NF + lprv] * sc + sb;
            float d = (l == 0) ? (prv - cur) : (cur - prv);
            B0t[jcol][kk] = (__bf16)cur;
            B1t[jcol][kk] = (__bf16)d;
        }
        __syncthreads();
        v16bf bb0[2], bb1[2];
#pragma unroll
        for (int ni = 0; ni < 2; ++ni) {
            bb0[ni] = load_bfrag(&B0t[ni * 16 + lm][16 * half]);
            bb1[ni] = load_bfrag(&B1t[ni * 16 + lm][16 * half]);
        }
#pragma unroll
        for (int mi = 0; mi < 2; ++mi) {
            int mt = wave + 8 * mi;
            v16bf a0 = *((const v16bf*)a0f + ((size_t)(mt * 8 + ks) * 32 + lane));
            v16bf a1 = *((const v16bf*)a1f + ((size_t)(mt * 8 + ks) * 32 + lane));
#pragma unroll
            for (int ni = 0; ni < 2; ++ni) {
                acc[mi][0][ni] = __builtin_amdgcn_wmma_f32_16x16x32_bf16(
                    false, a0, false, bb0[ni], (short)0, acc[mi][0][ni], false, false);
                acc[mi][1][ni] = __builtin_amdgcn_wmma_f32_16x16x32_bf16(
                    false, a1, false, bb1[ni], (short)0, acc[mi][1][ni], false, false);
            }
        }
        __syncthreads();
    }
    float* xo = xout + (size_t)r * RC;
#pragma unroll
    for (int ni = 0; ni < 2; ++ni) {
        int l = L0 + ni * 16 + lm;
        if (l >= NF) continue;
#pragma unroll
        for (int mi = 0; mi < 2; ++mi) {
            int m0 = (wave + 8 * mi) * 16;
#pragma unroll
            for (int g = 0; g < 8; ++g) {
                int o = m0 + g + 8 * half;
                float v0 = gelu_exact((acc[mi][0][ni][g] + b0[o]) * (g0[o] * BN_INV) + be0[o]);
                float v1 = gelu_exact((acc[mi][1][ni][g] + b1[o]) * (g1[o] * BN_INV) + be1[o]);
                xo[o * NF + l] = xr[o * NF + l] + v0 + v1;
            }
        }
    }
}

// ---------------------------------------------------------------------------
// K3: generic depthwise 3-tap conv (+bias), no activation
// ---------------------------------------------------------------------------
__global__ __launch_bounds__(256)
void dwconv_kernel(const float* __restrict__ in, const float* __restrict__ w,
                   const float* __restrict__ bias, float* __restrict__ out)
{
    int r = blockIdx.x;
    const float* src = in + (size_t)r * RC;
    float* dst = out + (size_t)r * RC;
    for (int idx = threadIdx.x; idx < RC; idx += blockDim.x) {
        int c = idx / NF, l = idx % NF;
        const float* wc = w + (size_t)c * 3;
        float xm = (l > 0)      ? src[c * NF + l - 1] : 0.f;
        float xc = src[c * NF + l];
        float xp = (l < NF - 1) ? src[c * NF + l + 1] : 0.f;
        dst[idx] = wc[0] * xm + wc[1] * xc + wc[2] * xp + bias[c];
    }
}

// ---------------------------------------------------------------------------
// K4: pointwise conv + BN + ReLU, WMMA bf16, N=32 per WG, swizzled bf16 A
// ---------------------------------------------------------------------------
__global__ __launch_bounds__(256)
void pwconv_bn_relu_kernel(const float* __restrict__ in, const __bf16* __restrict__ af,
                           const float* __restrict__ bias, const float* __restrict__ g,
                           const float* __restrict__ be, float* __restrict__ out)
{
    __shared__ __attribute__((aligned(32))) __bf16 Bt[32][40];   // [col][k]
    int r = blockIdx.y;
    int L0 = blockIdx.x * 32;
    int tid = threadIdx.x, lane = tid & 31, wave = tid >> 5;
    int lm = lane & 15, half = lane >> 4;
    const float* xr = in + (size_t)r * RC;

    v8f acc[2][2] = {};                       // [mi][ni]
    for (int ks = 0; ks < 8; ++ks) {
        int k0 = ks * 32;
#pragma unroll
        for (int u = 0; u < 4; ++u) {
            int idx = tid + u * 256;
            int jcol = idx & 31, kk = idx >> 5;
            int l = L0 + jcol; if (l > NF - 1) l = NF - 1;
            Bt[jcol][kk] = (__bf16)xr[(k0 + kk) * NF + l];
        }
        __syncthreads();
        v16bf bb[2];
#pragma unroll
        for (int ni = 0; ni < 2; ++ni)
            bb[ni] = load_bfrag(&Bt[ni * 16 + lm][16 * half]);
#pragma unroll
        for (int mi = 0; mi < 2; ++mi) {
            int mt = wave + 8 * mi;
            v16bf a = *((const v16bf*)af + ((size_t)(mt * 8 + ks) * 32 + lane));
#pragma unroll
            for (int ni = 0; ni < 2; ++ni)
                acc[mi][ni] = __builtin_amdgcn_wmma_f32_16x16x32_bf16(
                    false, a, false, bb[ni], (short)0, acc[mi][ni], false, false);
        }
        __syncthreads();
    }
    float* dst = out + (size_t)r * RC;
#pragma unroll
    for (int ni = 0; ni < 2; ++ni) {
        int l = L0 + ni * 16 + lm;
        if (l >= NF) continue;
#pragma unroll
        for (int mi = 0; mi < 2; ++mi) {
            int m0 = (wave + 8 * mi) * 16;
#pragma unroll
            for (int g2 = 0; g2 < 8; ++g2) {
                int o = m0 + g2 + 8 * half;
                float v = (acc[mi][ni][g2] + bias[o]) * (g[o] * BN_INV) + be[o];
                dst[o * NF + l] = fmaxf(v, 0.f);
            }
        }
    }
}

// ---------------------------------------------------------------------------
// K5: per-(b,m) cosine similarity S = F F^T, F = L2-normalized (100x256) rows.
// ---------------------------------------------------------------------------
__global__ __launch_bounds__(256)
void sim_kernel(const float* __restrict__ xcat, const float* __restrict__ xmix,
                __bf16* __restrict__ sim)
{
    int bm = blockIdx.x;
    int b = bm / 6, m = bm % 6;
    const float* src = (m < 3) ? (xcat + (size_t)(b * 3 + m) * RC)
                               : (xmix + (size_t)(b * 3 + m - 3) * RC);
    __shared__ __attribute__((aligned(32))) __bf16 Fb[112][264];
    __shared__ float rinv[112];
    int tid = threadIdx.x;

    if (tid < 112) {
        float s = 0.f;
        if (tid < NF)
            for (int c = 0; c < DIM; ++c) { float v = src[c * NF + tid]; s += v * v; }
        float n = fmaxf(sqrtf(s), 1e-8f);
        rinv[tid] = (tid < NF) ? (1.0f / n) : 0.f;
    }
    __syncthreads();
    for (int idx = tid; idx < 112 * DIM; idx += 256) {
        int l = idx >> 8, c = idx & 255;
        float v = (l < NF) ? src[c * NF + l] * rinv[l] : 0.f;
        Fb[l][c] = (__bf16)v;
    }
    __syncthreads();

    int lane = tid & 31, wave = tid >> 5, lm = lane & 15, half = lane >> 4;
    for (int t = wave; t < 49; t += 8) {
        int ti = t / 7, tj = t % 7;
        v8f acc = {};
        for (int k0 = 0; k0 < DIM; k0 += 32) {
            v16bf a, bb;
#pragma unroll
            for (int j = 0; j < 16; ++j) {
                a[j]  = Fb[ti * 16 + lm][k0 + a_perm(j, half)];
                bb[j] = Fb[tj * 16 + lm][k0 + 16 * half + j];
            }
            acc = __builtin_amdgcn_wmma_f32_16x16x32_bf16(
                false, a, false, bb, (short)0, acc, false, false);
        }
        int jj = tj * 16 + lm;
        if (jj < NF) {
#pragma unroll
            for (int g = 0; g < 8; ++g) {
                int ii = ti * 16 + g + 8 * half;
                if (ii < NF)
                    sim[(((size_t)b * 6 + m) * NF + ii) * NF + jj] = (__bf16)acc[g];
            }
        }
    }
}

// ---------------------------------------------------------------------------
// K6: templated 3x3 conv (pad 1), implicit GEMM, bf16 in, fused BN.
// Tap-major K, pre-swizzled bf16 A frags, bf16 n-major LDS B tile.
// MODE 0: ReLU + store bf16.  MODE 1: relu(resid + bn(conv)) -> atomic
// row-mean into mean_out[b,o,oy] (diff_map).
// ---------------------------------------------------------------------------
template<int Cin, int H, int W, int STRIDE, int Ho, int Wo, int Cout, int MODE>
__global__ __launch_bounds__(256)
void conv3x3_t_kernel(const __bf16* __restrict__ in, const __bf16* __restrict__ wfrag,
                      const float* __restrict__ bng, const float* __restrict__ bnb,
                      __bf16* __restrict__ outbf, const __bf16* __restrict__ resid,
                      float* __restrict__ mean_out)
{
    constexpr int Npix = Ho * Wo;
    constexpr int Ktot = Cin * 9;
    constexpr int KSTEPS = (Ktot + 31) / 32;
    constexpr int nm = Cout / 16;
    __shared__ __attribute__((aligned(32))) __bf16 Bt[32][40];   // [n][k]
    int b = blockIdx.y;
    int n0 = blockIdx.x * 32;
    int tid = threadIdx.x, lane = tid & 31, wave = tid >> 5;
    int lm = lane & 15, half = lane >> 4;
    const __bf16* inb = in + (size_t)b * Cin * H * W;

    v8f acc[2][2] = {};                       // [mi][ni]
    for (int ks = 0; ks < KSTEPS; ++ks) {
        int k0 = ks * 32;
#pragma unroll
        for (int u = 0; u < 4; ++u) {
            int idx = tid + u * 256;          // 32 n x 32 k
            int n = idx & 31, kk = idx >> 5;
            int k = k0 + kk;
            int npos = n0 + n;
            float v = 0.f;
            if (k < Ktot && npos < Npix) {
                int t = k / Cin;              // compile-time divisor
                int c = k - t * Cin;
                int oy = npos / Wo;           // compile-time divisor
                int ox = npos - oy * Wo;
                int iy = oy * STRIDE + (t / 3) - 1;
                int ix = ox * STRIDE + (t % 3) - 1;
                if (iy >= 0 && iy < H && ix >= 0 && ix < W)
                    v = (float)inb[(size_t)c * (H * W) + iy * W + ix];
            }
            Bt[n][kk] = (__bf16)v;
        }
        __syncthreads();
        v16bf bb[2];
#pragma unroll
        for (int ni = 0; ni < 2; ++ni)
            bb[ni] = load_bfrag(&Bt[ni * 16 + lm][16 * half]);
#pragma unroll
        for (int mi = 0; mi < 2; ++mi) {
            int mt = wave + 8 * mi;
            if (mt < nm) {
                v16bf a = *((const v16bf*)wfrag + ((size_t)(mt * KSTEPS + ks) * 32 + lane));
#pragma unroll
                for (int ni = 0; ni < 2; ++ni)
                    acc[mi][ni] = __builtin_amdgcn_wmma_f32_16x16x32_bf16(
                        false, a, false, bb[ni], (short)0, acc[mi][ni], false, false);
            }
        }
        __syncthreads();
    }
#pragma unroll
    for (int ni = 0; ni < 2; ++ni) {
        int npos = n0 + ni * 16 + lm;
        if (npos >= Npix) continue;
        int oy = npos / Wo;
#pragma unroll
        for (int mi = 0; mi < 2; ++mi) {
            int mt = wave + 8 * mi;
            if (mt >= nm) continue;
#pragma unroll
            for (int g = 0; g < 8; ++g) {
                int o = mt * 16 + g + 8 * half;
                size_t oidx = (size_t)b * Cout * Npix + (size_t)o * Npix + npos;
                float v = acc[mi][ni][g] * (bng[o] * BN_INV) + bnb[o];
                if (MODE == 0) {
                    outbf[oidx] = (__bf16)fmaxf(v, 0.f);
                } else {
                    v = fmaxf(v + (float)resid[oidx], 0.f);
                    atomicAdd(&mean_out[((size_t)b * Cout + o) * Ho + oy],
                              v * (1.0f / (float)Wo));
                }
            }
        }
    }
}

// ---------------------------------------------------------------------------
// small helpers
// ---------------------------------------------------------------------------
__global__ void zero_kernel(float* p, int n) {
    int i = blockIdx.x * blockDim.x + threadIdx.x;
    if (i < n) p[i] = 0.f;
}

__global__ __launch_bounds__(256)
void sd1d_kernel(const float* __restrict__ d1d, float* __restrict__ sd1) {
    int b = blockIdx.x, c = threadIdx.x;
    float s = 0.f;
    for (int i = 0; i < 3; ++i) {
        const float* p = d1d + ((size_t)(b * 3 + i) * DIM + c) * NF;
        for (int l = 0; l < NF; ++l) s += p[l];
    }
    sd1[b * DIM + c] = s * (1.0f / 300.0f);
}

__global__ __launch_bounds__(256)
void se_fuse_kernel(const float* __restrict__ sd1, const float* __restrict__ dmap,
                    const float* __restrict__ se1w1, const float* __restrict__ se1w2,
                    const float* __restrict__ se2w1, const float* __restrict__ se2w2,
                    const float* __restrict__ fw, const float* __restrict__ fb,
                    float* __restrict__ out)
{
    int b = blockIdx.x, t = threadIdx.x;
    __shared__ float sd[256], sm[256], hA[64], hB[64], hcat[512];
    sd[t] = sd1[b * DIM + t];
    {
        float s = 0.f;
        const float* dm = dmap + ((size_t)b * DIM + t) * 25;
        for (int i = 0; i < 25; ++i) s += dm[i];
        sm[t] = s * (1.0f / 25.0f);
    }
    __syncthreads();
    if (t < 64) {
        float a = 0.f, c2 = 0.f;
        for (int c = 0; c < DIM; ++c) {
            a  += sd[c] * se1w1[t * DIM + c];
            c2 += sm[c] * se2w1[t * DIM + c];
        }
        hA[t] = fmaxf(a, 0.f);
        hB[t] = fmaxf(c2, 0.f);
    }
    __syncthreads();
    {
        float a = 0.f, c2 = 0.f;
        for (int h = 0; h < 64; ++h) {
            a  += hA[h] * se1w2[t * 64 + h];
            c2 += hB[h] * se2w2[t * 64 + h];
        }
        float gate1 = 1.f / (1.f + expf(-a));
        float gate2 = 1.f / (1.f + expf(-c2));
        hcat[t]       = sm[t] * gate1;  // img2ddm
        hcat[256 + t] = sd[t] * gate2;  // ddm2img
    }
    __syncthreads();
    {
        float s = fb[t];
        for (int k = 0; k < 512; ++k) s += hcat[k] * fw[t * 512 + k];
        out[b * DIM + t] = fmaxf(s, 0.f);
    }
}

// ---------------------------------------------------------------------------
// host-side orchestration
// ---------------------------------------------------------------------------
extern "C" void kernel_launch(void* const* d_in, const int* in_sizes, int n_in,
                              void* d_out, int out_size, void* d_ws, size_t ws_size,
                              hipStream_t stream) {
    (void)in_sizes; (void)n_in; (void)out_size; (void)ws_size;
    const float* x0  = (const float*)d_in[0];
    const float* x1  = (const float*)d_in[1];
    const float* x2  = (const float*)d_in[2];
    const float* tw  = (const float*)d_in[3];
    const float* tb  = (const float*)d_in[4];
    const float* preg= (const float*)d_in[5];
    const float* preb= (const float*)d_in[6];
    const float* d0w = (const float*)d_in[7];
    const float* d0b = (const float*)d_in[8];
    const float* d0g = (const float*)d_in[9];
    const float* d0be= (const float*)d_in[10];
    const float* d1w = (const float*)d_in[11];
    const float* d1b = (const float*)d_in[12];
    const float* d1g = (const float*)d_in[13];
    const float* d1be= (const float*)d_in[14];
    const float* dedw= (const float*)d_in[15];
    const float* dedb= (const float*)d_in[16];
    const float* depw= (const float*)d_in[17];
    const float* depb= (const float*)d_in[18];
    const float* deg = (const float*)d_in[19];
    const float* debe= (const float*)d_in[20];
    const float* c1w = (const float*)d_in[21];
    const float* c1g = (const float*)d_in[22];
    const float* c1b = (const float*)d_in[23];
    const float* c2w = (const float*)d_in[24];
    const float* c2g = (const float*)d_in[25];
    const float* c2b = (const float*)d_in[26];
    const float* b1w = (const float*)d_in[27];
    const float* b1g = (const float*)d_in[28];
    const float* b1b = (const float*)d_in[29];
    const float* b2w = (const float*)d_in[30];
    const float* b2g = (const float*)d_in[31];
    const float* b2b = (const float*)d_in[32];
    const float* se1w1 = (const float*)d_in[33];
    const float* se1w2 = (const float*)d_in[34];
    const float* se2w1 = (const float*)d_in[35];
    const float* se2w2 = (const float*)d_in[36];
    const float* fw  = (const float*)d_in[37];
    const float* fbb = (const float*)d_in[38];

    char* ws = (char*)d_ws;
    const size_t OFF_XMIX = 78643200;   // 768*25600*4
    const size_t OFF_TMP  = 157286400;
    const size_t OFF_D1D  = 235929600;
    const size_t OFF_SIM  = 314572800;  // bf16 (256,6,100,100) = 30,720,000 B
    const size_t OFF_DMAP = 345292800;  // f32 (256,256,25) = 6,553,600 B
    const size_t OFF_SD1  = 351846400;  // f32 (256,256)
    // pre-swizzled bf16 weight fragments
    const size_t OFF_D0F  = 352108544;  // 131072 B
    const size_t OFF_D1F  = 352239616;  // 131072 B
    const size_t OFF_PW0F = 352370688;  // 131072 B
    const size_t OFF_PW1F = 352501760;  // 131072 B
    const size_t OFF_C1F  = 352632832;  // 8192 B
    const size_t OFF_C2F  = 352641024;  // 294912 B
    const size_t OFF_B1F  = 352935936;  // 1179648 B
    const size_t OFF_B2F  = 354115584;  // 1179648 B -> end ~355.3 MB
    const size_t OFF_H2   = 81920000;   // bf16 h2 after h1 ([0,235.9MB) free then)

    float* xcat = (float*)(ws);
    float* xmix = (float*)(ws + OFF_XMIX);
    float* tmp  = (float*)(ws + OFF_TMP);
    float* d1d  = (float*)(ws + OFF_D1D);
    __bf16* sim = (__bf16*)(ws + OFF_SIM);
    float* dmap = (float*)(ws + OFF_DMAP);
    float* sd1  = (float*)(ws + OFF_SD1);
    __bf16* d0f  = (__bf16*)(ws + OFF_D0F);
    __bf16* d1f  = (__bf16*)(ws + OFF_D1F);
    __bf16* pw0f = (__bf16*)(ws + OFF_PW0F);
    __bf16* pw1f = (__bf16*)(ws + OFF_PW1F);
    __bf16* c1f  = (__bf16*)(ws + OFF_C1F);
    __bf16* c2f  = (__bf16*)(ws + OFF_C2F);
    __bf16* b1f  = (__bf16*)(ws + OFF_B1F);
    __bf16* b2f  = (__bf16*)(ws + OFF_B2F);
    __bf16* h1  = (__bf16*)(ws);            // reuses xcat region (dead after sim)
    __bf16* h2  = (__bf16*)(ws + OFF_H2);
    __bf16* rbf = (__bf16*)(ws);            // reuses h1 region (dead after conv2)

    // 0) one-shot bf16 fragment pre-swizzles (tiny, deterministic each call)
    wfrag_gemm_kernel<<<64, 256, 0, stream>>>(d0w, d0f, 256, 256);
    wfrag_gemm_kernel<<<64, 256, 0, stream>>>(d1w, d1f, 256, 256);
    wfrag_gemm_kernel<<<64, 256, 0, stream>>>(depw, pw0f, 256, 256);
    wfrag_gemm_kernel<<<64, 256, 0, stream>>>(depw + 65536, pw1f, 256, 256);
    wfrag_conv_kernel<<<16, 256, 0, stream>>>(c1w, c1f, 64, 6);
    wfrag_conv_kernel<<<144, 256, 0, stream>>>(c2w, c2f, 256, 64);
    wfrag_conv_kernel<<<576, 256, 0, stream>>>(b1w, b1f, 256, 256);
    wfrag_conv_kernel<<<576, 256, 0, stream>>>(b2w, b2f, 256, 256);

    // 1) tconv + residual avg -> xcat (== out_list)
    tconv_kernel<<<768, 256, 0, stream>>>(x0, x1, x2, tw, tb, xcat);
    // 2) DiffMixer -> xmix
    diffmixer_kernel<<<dim3(4, 768), 256, 0, stream>>>(
        xcat, d0f, d0b, d0g, d0be, d1f, d1b, d1g, d1be, preg, preb, xmix);
    // 3-6) two diff-encoder blocks -> d1d
    dwconv_kernel<<<768, 256, 0, stream>>>(xmix, dedw, dedb, tmp);
    pwconv_bn_relu_kernel<<<dim3(4, 768), 256, 0, stream>>>(tmp, pw0f, depb, deg, debe, d1d);
    dwconv_kernel<<<768, 256, 0, stream>>>(d1d, dedw + 768, dedb + 256, tmp);
    pwconv_bn_relu_kernel<<<dim3(4, 768), 256, 0, stream>>>(
        tmp, pw1f, depb + 256, deg + 256, debe + 256, d1d);
    // 7) cosine similarity map (bf16)
    sim_kernel<<<1536, 256, 0, stream>>>(xcat, xmix, sim);
    // 8) zero diff_map accumulator
    zero_kernel<<<(1638400 + 255) / 256, 256, 0, stream>>>(dmap, 1638400);
    // 9) ResNet stem + basic block (implicit-GEMM WMMA convs)
    conv3x3_t_kernel<6, 100, 100, 2, 50, 50, 64, 0>
        <<<dim3(79, 256), 256, 0, stream>>>(sim, c1f, c1g, c1b, h1, nullptr, nullptr);
    conv3x3_t_kernel<64, 50, 50, 2, 25, 25, 256, 0>
        <<<dim3(20, 256), 256, 0, stream>>>(h1, c2f, c2g, c2b, h2, nullptr, nullptr);
    conv3x3_t_kernel<256, 25, 25, 1, 25, 25, 256, 0>
        <<<dim3(20, 256), 256, 0, stream>>>(h2, b1f, b1g, b1b, rbf, nullptr, nullptr);
    conv3x3_t_kernel<256, 25, 25, 1, 25, 25, 256, 1>
        <<<dim3(20, 256), 256, 0, stream>>>(rbf, b2f, b2g, b2b, nullptr, h2, dmap);
    // 10) pooled d1d
    sd1d_kernel<<<256, 256, 0, stream>>>(d1d, sd1);
    // 11) cross-SE + fuse -> d_out (flat (256,256) row-major == reshape(-1,256,8))
    se_fuse_kernel<<<256, 256, 0, stream>>>(
        sd1, dmap, se1w1, se1w2, se2w1, se2w2, fw, fbb, (float*)d_out);
}
